// FP6Linear_67242007986737
// MI455X (gfx1250) — matrix operand order repaired and checked by reading, not compiled, not run on hardware
//
#include <hip/hip_runtime.h>
#include <stdint.h>

// ---------------------------------------------------------------------------
// FP6Linear forward for MI455X (gfx1250, wave32, WMMA + async-to-LDS copies).
// out[M,N] = x[M,K] . dequant_fp6(W)[N,K]^T + bias
// M=8192, K=4096, N=16384 for the reference shapes. Compute-bound GEMM:
//   v_wmma_f32_16x16x32_bf16 for MACs, global_load_async_to_lds_b128 for
//   tile movement (ASYNCcnt-tracked, bypasses VGPR staging).
// ---------------------------------------------------------------------------

typedef __attribute__((ext_vector_type(16))) __bf16 v16bf;
typedef __attribute__((ext_vector_type(8)))  float  v8f;

#define TILE_M 128
#define TILE_N 128
#define TILE_K 32
#define LDSS   40   // padded LDS row stride (elements): 80B rows, bank-conflict free

// float -> bf16 round-to-nearest-even
__device__ __forceinline__ unsigned short f2bf(float f) {
    unsigned u = __float_as_uint(f);
    unsigned r = u + 0x7FFFu + ((u >> 16) & 1u);
    return (unsigned short)(r >> 16);
}

__global__ void init_amax(unsigned* amax) { *amax = 0u; }

// Pass 1: per-tensor abs-max of W. abs values are >= 0 so uint atomicMax works.
__global__ __launch_bounds__(256) void absmax_kernel(const float* __restrict__ W,
                                                     size_t n,
                                                     unsigned* __restrict__ amax) {
    __shared__ float red[256];
    float m = 0.f;
    for (size_t i = (size_t)blockIdx.x * blockDim.x + threadIdx.x; i < n;
         i += (size_t)gridDim.x * blockDim.x)
        m = fmaxf(m, fabsf(W[i]));
    red[threadIdx.x] = m;
    __syncthreads();
    for (int s = 128; s > 0; s >>= 1) {
        if ((int)threadIdx.x < s)
            red[threadIdx.x] = fmaxf(red[threadIdx.x], red[threadIdx.x + s]);
        __syncthreads();
    }
    if (threadIdx.x == 0) atomicMax(amax, __float_as_uint(red[0]));
}

__device__ __forceinline__ float fp6_qdq(float v, float scale, float inv_scale) {
    float s = fminf(fmaxf(v * inv_scale, -16.f), 16.f);
    float q = fminf(fmaxf(rintf((s + 16.f) * (63.f / 32.f)), 0.f), 63.f);
    return (q * (32.f / 63.f) - 16.f) * scale;
}

// Pass 2: quant-dequant W and store as bf16 (4 elements / thread-iter).
__global__ __launch_bounds__(256) void quant_w(const float4* __restrict__ W4,
                                               unsigned long long* __restrict__ Wq8,
                                               size_t n4,
                                               const unsigned* __restrict__ amax) {
    float am = __uint_as_float(*amax);
    float scale = am > 0.f ? am * (1.f / 16.f) : 1.f;
    float inv = 1.f / scale;
    for (size_t i = (size_t)blockIdx.x * blockDim.x + threadIdx.x; i < n4;
         i += (size_t)gridDim.x * blockDim.x) {
        float4 w = W4[i];
        unsigned long long p =
              (unsigned long long)f2bf(fp6_qdq(w.x, scale, inv))
            | ((unsigned long long)f2bf(fp6_qdq(w.y, scale, inv)) << 16)
            | ((unsigned long long)f2bf(fp6_qdq(w.z, scale, inv)) << 32)
            | ((unsigned long long)f2bf(fp6_qdq(w.w, scale, inv)) << 48);
        Wq8[i] = p;
    }
}

// Pass 3: x -> bf16
__global__ __launch_bounds__(256) void cvt_x(const float4* __restrict__ X4,
                                             unsigned long long* __restrict__ Xb8,
                                             size_t n4) {
    for (size_t i = (size_t)blockIdx.x * blockDim.x + threadIdx.x; i < n4;
         i += (size_t)gridDim.x * blockDim.x) {
        float4 v = X4[i];
        unsigned long long p =
              (unsigned long long)f2bf(v.x)
            | ((unsigned long long)f2bf(v.y) << 16)
            | ((unsigned long long)f2bf(v.z) << 32)
            | ((unsigned long long)f2bf(v.w) << 48);
        Xb8[i] = p;
    }
}

// 32B global -> LDS async copy (two b128, offset applies to BOTH addresses).
__device__ __forceinline__ void async_copy32(unsigned lds_addr,
                                             const unsigned short* ga) {
    unsigned long long g = (unsigned long long)(uintptr_t)ga;
    asm volatile("global_load_async_to_lds_b128 %0, %1, off\n\t"
                 "global_load_async_to_lds_b128 %0, %1, off offset:16"
                 :: "v"(lds_addr), "v"(g) : "memory");
}

__device__ __forceinline__ void wait_async0() {
    asm volatile("s_wait_asynccnt 0x0" ::: "memory");
}

// Pass 4: bf16 WMMA GEMM.  A = xb [M,K] row-major, B = Wq [N,K] row-major
// (W rows are K-contiguous, i.e. already the columns WMMA's B operand wants).
__global__ __launch_bounds__(256) void gemm_bf16_wmma(
    const unsigned short* __restrict__ A,
    const unsigned short* __restrict__ B,
    const float* __restrict__ bias,
    float* __restrict__ C,
    int M, int N, int K)
{
    __shared__ __align__(16) unsigned short As[2][TILE_M * LDSS];
    __shared__ __align__(16) unsigned short Bs[2][TILE_N * LDSS];

    const int tid  = threadIdx.x;
    const int lane = tid & 31;
    const int wave = tid >> 5;      // 0..7
    const int wm   = wave >> 1;     // 0..3 -> 32-row strip
    const int wn   = wave & 1;      // 0..1 -> 64-col strip
    const int half = lane >> 4;     // lane group (ISA 16-bit operand layout)
    const int l16  = lane & 15;

    const int tile_n = blockIdx.x * TILE_N;
    const int tile_m = blockIdx.y * TILE_M;

    // Tile fill: each of 256 threads async-copies 32B of A and 32B of B / K-step.
    const int lrow = tid >> 1;      // 0..127
    const int lhal = tid & 1;       // 16-element chunk of the K-slice
    const unsigned short* aptr = A + (size_t)(tile_m + lrow) * K + lhal * 16;
    const unsigned short* bptr = B + (size_t)(tile_n + lrow) * K + lhal * 16;

    // LDS destination byte addresses (flat shared addr[31:0] == LDS offset).
    unsigned ldsA[2], ldsB[2];
    ldsA[0] = (unsigned)(uintptr_t)&As[0][lrow * LDSS + lhal * 16];
    ldsA[1] = (unsigned)(uintptr_t)&As[1][lrow * LDSS + lhal * 16];
    ldsB[0] = (unsigned)(uintptr_t)&Bs[0][lrow * LDSS + lhal * 16];
    ldsB[1] = (unsigned)(uintptr_t)&Bs[1][lrow * LDSS + lhal * 16];

    v8f acc[2][4];
    v8f zero = {0.f, 0.f, 0.f, 0.f, 0.f, 0.f, 0.f, 0.f};
#pragma unroll
    for (int i = 0; i < 2; ++i)
#pragma unroll
        for (int j = 0; j < 4; ++j) acc[i][j] = zero;

    const int nk = K / TILE_K;

    // Prologue: async-fill buffer 0, publish to workgroup.
    async_copy32(ldsA[0], aptr);
    async_copy32(ldsB[0], bptr);
    wait_async0();
    __syncthreads();

    union Frag { uint4 u[2]; v16bf v; };

    for (int kt = 0; kt < nk; ++kt) {
        const int  cur = kt & 1;
        const bool have_next = (kt + 1) < nk;
        if (have_next) {            // overlap next tile's async copy with WMMAs
            const int nxt = cur ^ 1;
            async_copy32(ldsA[nxt], aptr + (size_t)(kt + 1) * TILE_K);
            async_copy32(ldsB[nxt], bptr + (size_t)(kt + 1) * TILE_K);
        }

        // A fragment: 16-bit A 16x32 layout — lane half 0: K 0..7 & 16..23,
        // lane half 1: K 8..15 & 24..31, row = l16.
        Frag af[2];
#pragma unroll
        for (int mi = 0; mi < 2; ++mi) {
            const unsigned short* p =
                &As[cur][(wm * 32 + mi * 16 + l16) * LDSS + half * 8];
            af[mi].u[0] = *(const uint4*)(p);
            af[mi].u[1] = *(const uint4*)(p + 16);
        }
        // B fragment: 16-bit B 32x16 layout — col = l16, lane half 0: K 0..7 & 16..23, lane half 1: K 8..15 & 24..31.
        Frag bfm[4];
#pragma unroll
        for (int ni = 0; ni < 4; ++ni) {
            const unsigned short* p =
                &Bs[cur][(wn * 64 + ni * 16 + l16) * LDSS + half * 8];
            bfm[ni].u[0] = *(const uint4*)(p);
            bfm[ni].u[1] = *(const uint4*)(p + 16);
        }

#pragma unroll
        for (int mi = 0; mi < 2; ++mi)
#pragma unroll
            for (int ni = 0; ni < 4; ++ni)
                acc[mi][ni] = __builtin_amdgcn_wmma_f32_16x16x32_bf16(
                    false, af[mi].v, false, bfm[ni].v,
                    (short)0, acc[mi][ni], false, false);

        if (have_next) wait_async0();  // own async writes done before barrier
        __syncthreads();               // publish buffer (kt+1)&1 to all waves
    }

    // Epilogue: C/D 16x16 f32 layout — VGPR v: M = v + 8*half, N = l16.
#pragma unroll
    for (int mi = 0; mi < 2; ++mi) {
#pragma unroll
        for (int ni = 0; ni < 4; ++ni) {
            const int n  = tile_n + wn * 64 + ni * 16 + l16;
            const float bv = bias[n];
            const int mb = tile_m + wm * 32 + mi * 16 + half * 8;
#pragma unroll
            for (int v = 0; v < 8; ++v)
                C[(size_t)(mb + v) * N + n] = acc[mi][ni][v] + bv;
        }
    }
}

extern "C" void kernel_launch(void* const* d_in, const int* in_sizes, int n_in,
                              void* d_out, int out_size, void* d_ws, size_t ws_size,
                              hipStream_t stream) {
    const float* x    = (const float*)d_in[0];
    const float* W    = (const float*)d_in[1];
    const float* bias = (const float*)d_in[2];
    float* out = (float*)d_out;

    const size_t xn = (size_t)in_sizes[0];   // M*K
    const size_t wn = (size_t)in_sizes[1];   // N*K
    const int N = in_sizes[2];
    const int K = (int)(wn / (size_t)N);
    const int M = (int)(xn / (size_t)K);

    // Workspace layout: [amax u32 | Wq bf16 (N*K) | xb bf16 (M*K)]
    unsigned* amax = (unsigned*)d_ws;
    unsigned short* Wq = (unsigned short*)((char*)d_ws + 256);
    unsigned short* xb = (unsigned short*)((char*)d_ws + 256 + wn * 2);

    init_amax<<<1, 1, 0, stream>>>(amax);
    absmax_kernel<<<2048, 256, 0, stream>>>(W, wn, amax);

    {
        size_t n4 = wn / 4;
        int blocks = (int)((n4 + 255) / 256);
        quant_w<<<blocks, 256, 0, stream>>>((const float4*)W,
                                            (unsigned long long*)Wq, n4, amax);
    }
    {
        size_t n4 = xn / 4;
        int blocks = (int)((n4 + 255) / 256);
        cvt_x<<<blocks, 256, 0, stream>>>((const float4*)x,
                                          (unsigned long long*)xb, n4);
    }

    dim3 grid(N / TILE_N, M / TILE_M);
    gemm_bf16_wmma<<<grid, 256, 0, stream>>>(xb, Wq, bias, out, M, N, K);
}
